// Siglip2Attention_80195629351436
// MI455X (gfx1250) — compile-verified, matrix operand order checked by reading, not run on hardware
//
#include <hip/hip_runtime.h>
#include <hip/hip_bf16.h>
#include <math.h>

// ---------------------------------------------------------------------------
// Siglip2-style document-masked attention for MI455X (gfx1250, wave32, WMMA)
// B=2, S=2048, D=1024, H=16, HD=64.  All GEMMs via v_wmma_f32_16x16x32_bf16.
// LDS-staged, double-buffered tiles; attention skips outside the document
// band (document_ids are sorted -> block-diagonal mask).
// ---------------------------------------------------------------------------

#define B_   2
#define S_   2048
#define D_   1024
#define H_   16
#define HD_  64
// softmax done in base-2 domain: scores * (HD^-0.5 * log2(e))
#define SCL2_ 0.1803368801111244f
#define NEGV_ (-1.0e30f)

typedef __attribute__((ext_vector_type(16))) __bf16 v16bf;
typedef __attribute__((ext_vector_type(8)))  __bf16 v8bf;
typedef __attribute__((ext_vector_type(4)))  __bf16 v4bf;
typedef __attribute__((ext_vector_type(8)))  float  v8f;

// A/B fragment loader for 16x16x32 bf16 WMMA.
// Per CDNA5 ISA: lane holds 8 contiguous bf16 at K-offset (lane>>4)*8 and
// another 8 at +16.  Caller passes p already offset by (lane>>4)*8.
__device__ __forceinline__ v16bf load_frag(const __bf16* p) {
  v8bf lo = *(const v8bf*)(p);
  v8bf hh = *(const v8bf*)(p + 16);
  v16bf r;
#pragma unroll
  for (int i = 0; i < 8; ++i) { r[i] = lo[i]; r[i + 8] = hh[i]; }
  return r;
}

__device__ __forceinline__ v8f wmma_bf16(v16bf a, v16bf b, v8f c) {
  return __builtin_amdgcn_wmma_f32_16x16x32_bf16(false, a, false, b,
                                                 (short)0, c, false, false);
}

__device__ __forceinline__ v8f zero8() {
  v8f z;
#pragma unroll
  for (int i = 0; i < 8; ++i) z[i] = 0.0f;
  return z;
}

// first index i in [0,S) with a[i] >= val  (a sorted ascending)
__device__ __forceinline__ int lb_lds(const int* a, int val) {
  int lo = 0, hi = S_;
  while (lo < hi) { int m = (lo + hi) >> 1; if (a[m] < val) lo = m + 1; else hi = m; }
  return lo;
}
// first index i in [0,S) with a[i] > val
__device__ __forceinline__ int ub_lds(const int* a, int val) {
  int lo = 0, hi = S_;
  while (lo < hi) { int m = (lo + hi) >> 1; if (a[m] <= val) lo = m + 1; else hi = m; }
  return lo;
}

// ---------------------------------------------------------------------------
// Kernel 1: fp32 -> bf16 conversion (x and the four weight matrices)
// ---------------------------------------------------------------------------
__global__ void __launch_bounds__(256) cvt_f32_bf16_kernel(
    const float* __restrict__ src, __bf16* __restrict__ dst, int n4) {
  int i = blockIdx.x * 256 + threadIdx.x;
  if (i >= n4) return;
  const float4 f = *(const float4*)(src + (size_t)i * 4);
  v4bf o;
  o[0] = (__bf16)f.x; o[1] = (__bf16)f.y; o[2] = (__bf16)f.z; o[3] = (__bf16)f.w;
  *(v4bf*)(dst + (size_t)i * 4) = o;
}

// Cooperative 64x32 bf16 weight tile copy: 256 threads x 16B.
__device__ __forceinline__ void stage_w64x32(const __bf16* __restrict__ w,
                                             __bf16* dst, int n0, int k0,
                                             int tid) {
  const int n  = tid >> 2;          // 0..63
  const int kk = (tid & 3) * 8;     // 0,8,16,24
  const v8bf t = *(const v8bf*)(w + (size_t)(n0 + n) * D_ + k0 + kk);
  *(v8bf*)(dst + n * 32 + kk) = t;
}

// ---------------------------------------------------------------------------
// Kernel 2: fused QKV projection.  Block = 128 rows x 64 cols; 8 waves each
// own a 16-row tile and share LDS-staged, double-buffered weight tiles.
//   Q,K stored [B,H,S,HD] bf16; V stored transposed [B,H,HD,S] bf16.
// ---------------------------------------------------------------------------
__global__ void __launch_bounds__(256) qkv_gemm_kernel(
    const __bf16* __restrict__ xb,
    const __bf16* __restrict__ wqb, const __bf16* __restrict__ wkb,
    const __bf16* __restrict__ wvb,
    const float* __restrict__ bq, const float* __restrict__ bk,
    const float* __restrict__ bv,
    __bf16* __restrict__ qo, __bf16* __restrict__ ko,
    __bf16* __restrict__ vto) {
  __shared__ __align__(16) __bf16 sW[2][3][64 * 32];   // 24 KB

  const int tid  = threadIdx.x;
  const int lane = tid & 31;
  const int wid  = tid >> 5;
  const int m0 = (blockIdx.x >> 4) * 128 + wid * 16;   // 32 M-blocks
  const int n0 = (blockIdx.x & 15) * 64;               // 16 N-groups
  const int r  = lane & 15;
  const int hi = lane >> 4;

  v8f aq[4], ak[4], av[4];
#pragma unroll
  for (int t = 0; t < 4; ++t) { aq[t] = zero8(); ak[t] = zero8(); av[t] = zero8(); }

  stage_w64x32(wqb, sW[0][0], n0, 0, tid);
  stage_w64x32(wkb, sW[0][1], n0, 0, tid);
  stage_w64x32(wvb, sW[0][2], n0, 0, tid);
  __syncthreads();

  const __bf16* xrow = xb + (size_t)(m0 + r) * D_ + hi * 8;
  for (int k0 = 0; k0 < D_; k0 += 32) {
    const int cur = (k0 >> 5) & 1;
    const int nxt = cur ^ 1;
    if (k0 + 32 < D_) {            // overlap next-tile staging with math
      stage_w64x32(wqb, sW[nxt][0], n0, k0 + 32, tid);
      stage_w64x32(wkb, sW[nxt][1], n0, k0 + 32, tid);
      stage_w64x32(wvb, sW[nxt][2], n0, k0 + 32, tid);
      __builtin_prefetch(xrow + k0 + 32, 0, 0);        // global_prefetch_b8
    }
    const v16bf a = load_frag(xrow + k0);
    v16bf fq[4], fk[4], fv[4];
#pragma unroll
    for (int t = 0; t < 4; ++t) {
      const int off = (t * 16 + r) * 32 + hi * 8;
      fq[t] = load_frag(sW[cur][0] + off);
      fk[t] = load_frag(sW[cur][1] + off);
      fv[t] = load_frag(sW[cur][2] + off);
    }
#pragma unroll
    for (int t = 0; t < 4; ++t) {
      aq[t] = wmma_bf16(a, fq[t], aq[t]);
      ak[t] = wmma_bf16(a, fk[t], ak[t]);
      av[t] = wmma_bf16(a, fv[t], av[t]);
    }
    __syncthreads();
  }

  const int b  = m0 >> 11;        // m0 / S
  const int s0 = m0 & (S_ - 1);
#pragma unroll
  for (int t = 0; t < 4; ++t) {
    const int n = n0 + t * 16 + r;
    const int h = n >> 6, d = n & 63;
    const float biq = bq[n], bik = bk[n], biv = bv[n];
    const size_t qkbase = ((size_t)(b * H_ + h) * S_) * HD_ + d;
#pragma unroll
    for (int v = 0; v < 8; ++v) {
      const int s = s0 + v + 8 * hi;                      // C row M = v + 8*hi
      qo[qkbase + (size_t)s * HD_] = (__bf16)(aq[t][v] + biq);
      ko[qkbase + (size_t)s * HD_] = (__bf16)(ak[t][v] + bik);
    }
    v8bf pv;
#pragma unroll
    for (int v = 0; v < 8; ++v) pv[v] = (__bf16)(av[t][v] + biv);
    *(v8bf*)(vto + ((size_t)(b * H_ + h) * HD_ + d) * S_ + s0 + 8 * hi) = pv;
  }
}

// ---------------------------------------------------------------------------
// Kernel 3: flash attention with document-block mask.
// Block = one (b,h) x 128 queries (8 waves x 16).  document_ids are sorted,
// so each block only visits the key band covered by its queries' documents;
// each wave further skips tiles outside its own 16-query doc range.
// ---------------------------------------------------------------------------
__global__ void __launch_bounds__(256) attn_kernel(
    const __bf16* __restrict__ qo, const __bf16* __restrict__ ko,
    const __bf16* __restrict__ vto, const int* __restrict__ doc,
    __bf16* __restrict__ ctx) {
  __shared__ int sdoc[S_];                               // 8 KB
  __shared__ __align__(16) __bf16 sP[8 * 16 * 32];       // 8 KB, per-wave P
  __shared__ __align__(16) __bf16 sK[2][32 * 64];        // 8 KB
  __shared__ __align__(16) __bf16 sV[2][64 * 32];        // 8 KB

  const int tid  = threadIdx.x;
  const int lane = tid & 31;
  const int wid  = tid >> 5;

  for (int i = tid; i < S_; i += 256) sdoc[i] = doc[i];

  const int bh   = blockIdx.x >> 4;               // (b*H + h), 32 values
  const int qblk = (blockIdx.x & 15) * 128;
  const int q0   = qblk + wid * 16;
  const int b = bh >> 4, h = bh & 15;
  const int r = lane & 15, hi = lane >> 4;

  const __bf16* qp = qo  + (size_t)bh * S_ * HD_;
  const __bf16* kp = ko  + (size_t)bh * S_ * HD_;
  const __bf16* vp = vto + (size_t)bh * HD_ * S_;

  __syncthreads();   // sdoc ready

  // key band for the whole block (uniform) and for this wave
  const int blk_lo = lb_lds(sdoc, sdoc[qblk]);
  const int blk_hi = ub_lds(sdoc, sdoc[qblk + 127]);
  const int w_lo   = lb_lds(sdoc, sdoc[q0]);
  const int w_hi   = ub_lds(sdoc, sdoc[q0 + 15]);
  const int kstart = blk_lo & ~31;

  // stage first K/V tile (keys kstart..kstart+31)
  {
    const int key = tid >> 3, dk = (tid & 7) * 8;        // K: 32x64
    *(v8bf*)(sK[0] + key * 64 + dk) =
        *(const v8bf*)(kp + (size_t)(kstart + key) * HD_ + dk);
    const int dv = tid >> 2, ck = (tid & 3) * 8;         // V: 64x32
    *(v8bf*)(sV[0] + dv * 32 + ck) =
        *(const v8bf*)(vp + (size_t)dv * S_ + kstart + ck);
  }

  const v16bf a0 = load_frag(qp + (size_t)(q0 + r) * HD_ + hi * 8);       // d 0..31
  const v16bf a1 = load_frag(qp + (size_t)(q0 + r) * HD_ + 32 + hi * 8);  // d 32..63

  int docq[8];
#pragma unroll
  for (int v = 0; v < 8; ++v) docq[v] = sdoc[q0 + v + 8 * hi];

  float mrun[8], lrun[8];
#pragma unroll
  for (int v = 0; v < 8; ++v) { mrun[v] = -3.0e38f; lrun[v] = 0.0f; }
  v8f acc[4];
#pragma unroll
  for (int t = 0; t < 4; ++t) acc[t] = zero8();

  __bf16* Pl = sP + wid * (16 * 32);

  __syncthreads();   // first K/V tile ready

  for (int kt = kstart, it = 0; kt < blk_hi; kt += 32, ++it) {
    const int cur = it & 1;
    const int nxt = cur ^ 1;
    if (kt + 32 < blk_hi) {      // overlap next K/V staging with math
      const int key = tid >> 3, dk = (tid & 7) * 8;
      *(v8bf*)(sK[nxt] + key * 64 + dk) =
          *(const v8bf*)(kp + (size_t)(kt + 32 + key) * HD_ + dk);
      const int dv = tid >> 2, ck = (tid & 3) * 8;
      *(v8bf*)(sV[nxt] + dv * 32 + ck) =
          *(const v8bf*)(vp + (size_t)dv * S_ + kt + 32 + ck);
    }

    // wave-uniform skip: does this 32-key tile intersect this wave's band?
    if (kt + 32 > w_lo && kt < w_hi) {
      // scores: two 16x16 tiles (keys r and 16+r), K-dim = 64 (2 WMMAs each)
      v8f s0 = zero8(), s1 = zero8();
      {
        const v16bf b00 = load_frag(sK[cur] + r * 64 + hi * 8);
        const v16bf b01 = load_frag(sK[cur] + r * 64 + 32 + hi * 8);
        const v16bf b10 = load_frag(sK[cur] + (16 + r) * 64 + hi * 8);
        const v16bf b11 = load_frag(sK[cur] + (16 + r) * 64 + 32 + hi * 8);
        s0 = wmma_bf16(a0, b00, s0);
        s0 = wmma_bf16(a1, b01, s0);
        s1 = wmma_bf16(a0, b10, s1);
        s1 = wmma_bf16(a1, b11, s1);
      }

      const int dk0 = sdoc[kt + r], dk1 = sdoc[kt + 16 + r];
#pragma unroll
      for (int v = 0; v < 8; ++v) {
        // base-2 softmax domain: score * (HD^-0.5 * log2 e)
        float x0 = (docq[v] == dk0) ? s0[v] * SCL2_ : NEGV_;
        float x1 = (docq[v] == dk1) ? s1[v] * SCL2_ : NEGV_;
        float t = fmaxf(x0, x1);
        t = fmaxf(t, __shfl_xor(t, 1));
        t = fmaxf(t, __shfl_xor(t, 2));
        t = fmaxf(t, __shfl_xor(t, 4));
        t = fmaxf(t, __shfl_xor(t, 8));
        const float mn = fmaxf(mrun[v], t);
        const float p0 = exp2f(x0 - mn);
        const float p1 = exp2f(x1 - mn);
        float rs = p0 + p1;
        rs += __shfl_xor(rs, 1);
        rs += __shfl_xor(rs, 2);
        rs += __shfl_xor(rs, 4);
        rs += __shfl_xor(rs, 8);
        const float corr = exp2f(mrun[v] - mn);
        lrun[v] = lrun[v] * corr + rs;
        mrun[v] = mn;
#pragma unroll
        for (int t4 = 0; t4 < 4; ++t4) acc[t4][v] *= corr;
        const int row = v + 8 * hi;
        Pl[row * 32 + r]      = (__bf16)p0;
        Pl[row * 32 + 16 + r] = (__bf16)p1;
      }
      // LDS ops are in-order within a wave (DScnt): safe to read back.
      const v16bf pa = load_frag(Pl + r * 32 + hi * 8);
      v16bf fv[4];
#pragma unroll
      for (int t4 = 0; t4 < 4; ++t4)
        fv[t4] = load_frag(sV[cur] + (t4 * 16 + r) * 32 + hi * 8);
#pragma unroll
      for (int t4 = 0; t4 < 4; ++t4)
        acc[t4] = wmma_bf16(pa, fv[t4], acc[t4]);
    }

    __syncthreads();   // block-uniform: buffer handoff
  }

  // normalize and store context [B,S,D] bf16
#pragma unroll
  for (int t4 = 0; t4 < 4; ++t4) {
#pragma unroll
    for (int v = 0; v < 8; ++v) {
      const int s = q0 + v + 8 * hi;
      const float o = acc[t4][v] / lrun[v];
      ctx[((size_t)b * S_ + s) * D_ + h * HD_ + t4 * 16 + r] = (__bf16)o;
    }
  }
}

// ---------------------------------------------------------------------------
// Kernel 4: output projection, fp32 result.  Same LDS-staged structure.
// ---------------------------------------------------------------------------
__global__ void __launch_bounds__(256) out_gemm_kernel(
    const __bf16* __restrict__ cb, const __bf16* __restrict__ wob,
    const float* __restrict__ bo, float* __restrict__ out) {
  __shared__ __align__(16) __bf16 sW[2][64 * 32];        // 8 KB

  const int tid  = threadIdx.x;
  const int lane = tid & 31;
  const int wid  = tid >> 5;
  const int m0 = (blockIdx.x >> 4) * 128 + wid * 16;
  const int n0 = (blockIdx.x & 15) * 64;
  const int r  = lane & 15;
  const int hi = lane >> 4;

  v8f acc[4];
#pragma unroll
  for (int t = 0; t < 4; ++t) acc[t] = zero8();

  stage_w64x32(wob, sW[0], n0, 0, tid);
  __syncthreads();

  const __bf16* crow = cb + (size_t)(m0 + r) * D_ + hi * 8;
  for (int k0 = 0; k0 < D_; k0 += 32) {
    const int cur = (k0 >> 5) & 1;
    const int nxt = cur ^ 1;
    if (k0 + 32 < D_) {
      stage_w64x32(wob, sW[nxt], n0, k0 + 32, tid);
      __builtin_prefetch(crow + k0 + 32, 0, 0);
    }
    const v16bf a = load_frag(crow + k0);
    v16bf fw[4];
#pragma unroll
    for (int t = 0; t < 4; ++t)
      fw[t] = load_frag(sW[cur] + (t * 16 + r) * 32 + hi * 8);
#pragma unroll
    for (int t = 0; t < 4; ++t)
      acc[t] = wmma_bf16(a, fw[t], acc[t]);
    __syncthreads();
  }

#pragma unroll
  for (int t = 0; t < 4; ++t) {
    const int n = n0 + t * 16 + r;
    const float bias = bo[n];
#pragma unroll
    for (int v = 0; v < 8; ++v)
      out[(size_t)(m0 + v + 8 * hi) * D_ + n] = acc[t][v] + bias;
  }
}

// ---------------------------------------------------------------------------
// Launch
// ---------------------------------------------------------------------------
extern "C" void kernel_launch(void* const* d_in, const int* in_sizes, int n_in,
                              void* d_out, int out_size, void* d_ws, size_t ws_size,
                              hipStream_t stream) {
  const float* x  = (const float*)d_in[0];
  const float* wq = (const float*)d_in[1];
  const float* bq = (const float*)d_in[2];
  const float* wk = (const float*)d_in[3];
  const float* bk = (const float*)d_in[4];
  const float* wv = (const float*)d_in[5];
  const float* bv = (const float*)d_in[6];
  const float* wo = (const float*)d_in[7];
  const float* bo = (const float*)d_in[8];
  const int*  doc = (const int*)d_in[9];
  float* out = (float*)d_out;

  // workspace carve-out (bytes); total 48 MiB
  char* ws = (char*)d_ws;
  __bf16* xb  = (__bf16*)(ws + 0);          //  8 MiB  x   bf16 [B*S, D]
  __bf16* wqb = (__bf16*)(ws + 8388608);    //  2 MiB
  __bf16* wkb = (__bf16*)(ws + 10485760);   //  2 MiB
  __bf16* wvb = (__bf16*)(ws + 12582912);   //  2 MiB
  __bf16* wob = (__bf16*)(ws + 14680064);   //  2 MiB
  __bf16* qo  = (__bf16*)(ws + 16777216);   //  8 MiB  [B,H,S,HD]
  __bf16* ko  = (__bf16*)(ws + 25165824);   //  8 MiB  [B,H,S,HD]
  __bf16* vto = (__bf16*)(ws + 33554432);   //  8 MiB  [B,H,HD,S]
  __bf16* ctx = (__bf16*)(ws + 41943040);   //  8 MiB  [B,S,D]

  const int nX = B_ * S_ * D_;   // 4194304
  const int nW = D_ * D_;        // 1048576

  cvt_f32_bf16_kernel<<<(nX / 4 + 255) / 256, 256, 0, stream>>>(x,  xb,  nX / 4);
  cvt_f32_bf16_kernel<<<(nW / 4 + 255) / 256, 256, 0, stream>>>(wq, wqb, nW / 4);
  cvt_f32_bf16_kernel<<<(nW / 4 + 255) / 256, 256, 0, stream>>>(wk, wkb, nW / 4);
  cvt_f32_bf16_kernel<<<(nW / 4 + 255) / 256, 256, 0, stream>>>(wv, wvb, nW / 4);
  cvt_f32_bf16_kernel<<<(nW / 4 + 255) / 256, 256, 0, stream>>>(wo, wob, nW / 4);

  // 32 M-blocks (128 rows) x 16 N-groups (64 cols) = 512 blocks
  qkv_gemm_kernel<<<512, 256, 0, stream>>>(xb, wqb, wkb, wvb, bq, bk, bv,
                                           qo, ko, vto);
  // 32 (b,h) x 16 query-blocks (128 queries) = 512 blocks
  attn_kernel<<<512, 256, 0, stream>>>(qo, ko, vto, doc, ctx);

  out_gemm_kernel<<<512, 256, 0, stream>>>(ctx, wob, bo, out);
}
